// SimplifiedAttention_31714038513932
// MI455X (gfx1250) — matrix operand order repaired and checked by reading, not compiled, not run on hardware
//
#include <hip/hip_runtime.h>
#include <hip/hip_bf16.h>

typedef __attribute__((ext_vector_type(16))) __bf16 bf16x16;
typedef __attribute__((ext_vector_type(8)))  float  f32x8;

union Frag { bf16x16 v; uint4 q[2]; };

__device__ __forceinline__ f32x8 zero_f32x8() {
  f32x8 z;
#pragma unroll
  for (int i = 0; i < 8; ++i) z[i] = 0.0f;
  return z;
}

constexpr int cB  = 2;
constexpr int cS  = 2048;
constexpr int cH  = 1024;
constexpr int cNH = 16;
constexpr int cHD = 64;
constexpr int cM  = cB * cS;   // 4096
constexpr int c3H = 3 * cH;    // 3072

// ---------------------------------------------------------------------------
// Async global->LDS copy path (gfx1250). ASYNCcnt-tracked, bypasses VGPRs.
// Builtin prototype (from compiler diagnostic): (V4i addrspace(1)*,
// V4i addrspace(3)*, imm offset, imm cpol). Guarded by __has_builtin so the
// host pass / older toolchains fall back to a synchronous VGPR-bounce copy.
// ---------------------------------------------------------------------------
#if defined(__has_builtin)
#  if __has_builtin(__builtin_amdgcn_global_load_async_to_lds_b128) && \
      __has_builtin(__builtin_amdgcn_s_wait_asynccnt)
#    define USE_ASYNC_LDS 1
#  endif
#endif
#ifndef USE_ASYNC_LDS
#  define USE_ASYNC_LDS 0
#endif

#if USE_ASYNC_LDS
typedef int v4i_async __attribute__((ext_vector_type(4)));
typedef __attribute__((address_space(1))) v4i_async* async_gptr;
typedef __attribute__((address_space(3))) v4i_async* async_lptr;
#endif

__device__ __forceinline__ void tile_copy_b128(void* lds_dst, const void* gsrc) {
#if USE_ASYNC_LDS
  __builtin_amdgcn_global_load_async_to_lds_b128(
      (async_gptr)gsrc, (async_lptr)lds_dst, 0, 0);
#else
  *(uint4*)lds_dst = *(const uint4*)gsrc;
#endif
}

__device__ __forceinline__ void tile_copy_wait() {
#if USE_ASYNC_LDS
  __builtin_amdgcn_s_wait_asynccnt(0);
#endif
}

// ---------------------------------------------------------------------------
// Prep kernel 0a: f32 -> bf16
// ---------------------------------------------------------------------------
__global__ __launch_bounds__(256) void cvt_bf16(const float* __restrict__ s,
                                                __bf16* __restrict__ d)
{
  size_t i = ((size_t)blockIdx.x * 256 + threadIdx.x) * 4;
  float4 f = *(const float4*)(s + i);
  alignas(8) __bf16 t4[4] = {(__bf16)f.x, (__bf16)f.y, (__bf16)f.z, (__bf16)f.w};
  *(uint2*)(d + i) = *(const uint2*)t4;
}

// ---------------------------------------------------------------------------
// Prep kernel 0b: f32 [R][C] -> bf16 transposed [C][R] (weights become K-major
// so GEMM B-tiles are raw contiguous copies -> async-LDS eligible).
// ---------------------------------------------------------------------------
__global__ __launch_bounds__(256) void cvt_t_bf16(const float* __restrict__ s,
                                                  __bf16* __restrict__ d,
                                                  int srows, int scols)
{
  __shared__ float tile[32][33];
  const int tx = threadIdx.x & 31, ty = threadIdx.x >> 5;   // ty 0..7
  const int r0 = blockIdx.y * 32, c0 = blockIdx.x * 32;
#pragma unroll
  for (int i = 0; i < 4; ++i)
    tile[ty + i * 8][tx] = s[(size_t)(r0 + ty + i * 8) * scols + c0 + tx];
  __syncthreads();
#pragma unroll
  for (int i = 0; i < 4; ++i)
    d[(size_t)(c0 + ty + i * 8) * srows + r0 + tx] = (__bf16)tile[tx][ty + i * 8];
}

// ---------------------------------------------------------------------------
// Kernel 1: QKV GEMM, 128x128 tile, K-step 32, double-buffered async LDS.
// A = x_bf16 [4096][1024], B = WqkvT_bf16 [3072][1024] (K-major).
// Q,K written [B][NH][S][HD]; V written transposed [B][NH][HD][S].
// ---------------------------------------------------------------------------
__global__ __launch_bounds__(256) void qkv_gemm(
    const __bf16* __restrict__ xb, const __bf16* __restrict__ WqkvT,
    const float* __restrict__ bqkv,
    __bf16* __restrict__ Q, __bf16* __restrict__ K, __bf16* __restrict__ Vt)
{
  __shared__ __align__(16) __bf16 As[2][128][40];   // [m][k]
  __shared__ __align__(16) __bf16 Bt[2][128][40];   // [n][k]

  const int tid  = threadIdx.x;
  const int wave = tid >> 5, lane = tid & 31;
  const int hh   = lane >> 4, l16 = lane & 15;
  const int wm   = wave >> 1, wn  = wave & 1;
  const int m0   = blockIdx.x * 128, n0 = blockIdx.y * 128;

  auto load_tiles = [&](int buf, int k0) {
#pragma unroll
    for (int i = 0; i < 2; ++i) {                   // 512 b128 slots / 256 thr
      int idx = tid + i * 256;
      int r = idx >> 2, c8 = idx & 3;
      tile_copy_b128(&As[buf][r][c8 * 8], xb + (size_t)(m0 + r) * cH + k0 + c8 * 8);
      tile_copy_b128(&Bt[buf][r][c8 * 8], WqkvT + (size_t)(n0 + r) * cH + k0 + c8 * 8);
    }
  };

  f32x8 acc[2][4];
#pragma unroll
  for (int t = 0; t < 2; ++t)
#pragma unroll
    for (int u = 0; u < 4; ++u) acc[t][u] = zero_f32x8();

  load_tiles(0, 0);
  tile_copy_wait();
  __syncthreads();

  for (int kt = 0; kt < cH / 32; ++kt) {
    const int buf = kt & 1;
    if (kt + 1 < cH / 32) load_tiles(buf ^ 1, (kt + 1) * 32);  // overlap w/ WMMA

    Frag a[2], bb[4];
#pragma unroll
    for (int t = 0; t < 2; ++t) {
      int row = wm * 32 + t * 16 + l16;
      a[t].q[0] = *(const uint4*)&As[buf][row][hh * 8];
      a[t].q[1] = *(const uint4*)&As[buf][row][16 + hh * 8];
    }
#pragma unroll
    for (int u = 0; u < 4; ++u) {
      int col = wn * 64 + u * 16 + l16;
      bb[u].q[0] = *(const uint4*)&Bt[buf][col][hh * 8];
      bb[u].q[1] = *(const uint4*)&Bt[buf][col][16 + hh * 8];
    }
#pragma unroll
    for (int t = 0; t < 2; ++t)
#pragma unroll
      for (int u = 0; u < 4; ++u)
        acc[t][u] = __builtin_amdgcn_wmma_f32_16x16x32_bf16(
            false, a[t].v, false, bb[u].v, (short)0, acc[t][u], false, false);

    tile_copy_wait();
    __syncthreads();
  }

#pragma unroll
  for (int t = 0; t < 2; ++t)
#pragma unroll
    for (int u = 0; u < 4; ++u) {
      int n_g = n0 + wn * 64 + u * 16 + l16;
      int p = n_g >> 10, within = n_g & 1023;
      int head = within >> 6, d = within & 63;
      float bias = bqkv[n_g];
      if (p == 2) {                                  // V: [B][NH][HD][S]
#pragma unroll
        for (int r = 0; r < 8; ++r) {
          int m_g = m0 + wm * 32 + t * 16 + r + 8 * hh;
          int bi = m_g >> 11, si = m_g & 2047;
          Vt[(((size_t)bi * cNH + head) * cHD + d) * cS + si] =
              (__bf16)(acc[t][u][r] + bias);
        }
      } else {                                       // Q/K: [B][NH][S][HD]
        __bf16* dst = (p == 0) ? Q : K;
#pragma unroll
        for (int r = 0; r < 8; ++r) {
          int m_g = m0 + wm * 32 + t * 16 + r + 8 * hh;
          int bi = m_g >> 11, si = m_g & 2047;
          dst[(((size_t)bi * cNH + head) * cS + si) * cHD + d] =
              (__bf16)(acc[t][u][r] + bias);
        }
      }
    }
}

// ---------------------------------------------------------------------------
// Kernel 2: flash attention, block = (b, h, 64-query tile), 4 waves x 16 rows.
// K/V tiles double-buffered via async LDS; V is pre-transposed in global.
// ---------------------------------------------------------------------------
__global__ __launch_bounds__(128) void flash_attn(
    const __bf16* __restrict__ Q, const __bf16* __restrict__ K,
    const __bf16* __restrict__ Vt, const int* __restrict__ mask,
    __bf16* __restrict__ attended)
{
  __shared__ __align__(16) __bf16 Qs[64][80];        // [q][d]
  __shared__ __align__(16) __bf16 Ks[2][64][80];     // [key][d]
  __shared__ __align__(16) __bf16 Vs[2][64][80];     // [d][key]
  __shared__ __align__(16) __bf16 Ps[4][16][80];     // per-wave P [row][key]

  const int tid  = threadIdx.x;
  const int wave = tid >> 5, lane = tid & 31;
  const int hh   = lane >> 4, l16 = lane & 15;
  const int q0   = blockIdx.x * 64;
  const int h    = blockIdx.y;
  const int b    = blockIdx.z;

  const __bf16* Qb = Q  + ((size_t)b * cNH + h) * cS * cHD;
  const __bf16* Kb = K  + ((size_t)b * cNH + h) * cS * cHD;
  const __bf16* Vb = Vt + ((size_t)b * cNH + h) * cHD * cS;

  auto load_kv = [&](int buf, int kb) {
#pragma unroll
    for (int i = 0; i < 4; ++i) {                    // 512 b128 slots / 128 thr
      int idx = tid + i * 128;
      int r = idx >> 3, c8 = idx & 7;
      tile_copy_b128(&Ks[buf][r][c8 * 8], Kb + (size_t)(kb + r) * cHD + c8 * 8);
      tile_copy_b128(&Vs[buf][r][c8 * 8], Vb + (size_t)r * cS + kb + c8 * 8);
    }
  };

#pragma unroll
  for (int i = 0; i < 4; ++i) {                      // Q tile 64x64
    int idx = tid + i * 128;
    int r = idx >> 3, c8 = idx & 7;
    tile_copy_b128(&Qs[r][c8 * 8], Qb + (size_t)(q0 + r) * cHD + c8 * 8);
  }
  load_kv(0, 0);
  tile_copy_wait();
  __syncthreads();

  Frag aq[2];                                        // Q A-frags, loop-invariant
#pragma unroll
  for (int ks = 0; ks < 2; ++ks) {
    int row = wave * 16 + l16;
    aq[ks].q[0] = *(const uint4*)&Qs[row][ks * 32 + hh * 8];
    aq[ks].q[1] = *(const uint4*)&Qs[row][ks * 32 + 16 + hh * 8];
  }

  float mrow[8], lrow[8];
  f32x8 o[4];
#pragma unroll
  for (int r = 0; r < 8; ++r) { mrow[r] = -1.0e30f; lrow[r] = 0.0f; }
#pragma unroll
  for (int u = 0; u < 4; ++u) o[u] = zero_f32x8();

  for (int kt = 0; kt < cS / 64; ++kt) {
    const int buf = kt & 1;
    const int kb  = kt * 64;
    if (kt + 1 < cS / 64) load_kv(buf ^ 1, kb + 64); // overlap w/ compute

    f32x8 sacc[4];
#pragma unroll
    for (int u = 0; u < 4; ++u) sacc[u] = zero_f32x8();
#pragma unroll
    for (int ks = 0; ks < 2; ++ks)
#pragma unroll
      for (int u = 0; u < 4; ++u) {
        Frag bk;                                     // kdim=d contiguous in Ks row
        bk.q[0] = *(const uint4*)&Ks[buf][u * 16 + l16][ks * 32 + hh * 8];
        bk.q[1] = *(const uint4*)&Ks[buf][u * 16 + l16][ks * 32 + 16 + hh * 8];
        sacc[u] = __builtin_amdgcn_wmma_f32_16x16x32_bf16(
            false, aq[ks].v, false, bk.v, (short)0, sacc[u], false, false);
      }

    float sc[4][8], madd[4];
#pragma unroll
    for (int u = 0; u < 4; ++u) {
      int key = kb + u * 16 + l16;
      madd[u] = (mask[b * cS + key] == 0) ? -1.0e9f : 0.0f;
#pragma unroll
      for (int r = 0; r < 8; ++r) sc[u][r] = sacc[u][r] * 0.125f + madd[u];
    }
    float rm[8];
#pragma unroll
    for (int r = 0; r < 8; ++r) {                    // row-max over 16-lane group
      float m = fmaxf(fmaxf(sc[0][r], sc[1][r]), fmaxf(sc[2][r], sc[3][r]));
#pragma unroll
      for (int off = 8; off >= 1; off >>= 1) m = fmaxf(m, __shfl_xor(m, off, 32));
      rm[r] = m;
    }
    float alpha[8];
#pragma unroll
    for (int r = 0; r < 8; ++r) {
      float mn = fmaxf(mrow[r], rm[r]);
      alpha[r] = __expf(mrow[r] - mn);
      mrow[r] = mn;
    }
    float pf[4][8];
#pragma unroll
    for (int u = 0; u < 4; ++u)
#pragma unroll
      for (int r = 0; r < 8; ++r) pf[u][r] = __expf(sc[u][r] - mrow[r]);
#pragma unroll
    for (int r = 0; r < 8; ++r) {
      float s = pf[0][r] + pf[1][r] + pf[2][r] + pf[3][r];
#pragma unroll
      for (int off = 8; off >= 1; off >>= 1) s += __shfl_xor(s, off, 32);
      lrow[r] = lrow[r] * alpha[r] + s;
    }
#pragma unroll
    for (int u = 0; u < 4; ++u)
#pragma unroll
      for (int r = 0; r < 8; ++r) o[u][r] *= alpha[r];

    // P: C-layout regs -> A-layout via per-wave LDS (DS in-order within wave)
#pragma unroll
    for (int u = 0; u < 4; ++u)
#pragma unroll
      for (int r = 0; r < 8; ++r)
        Ps[wave][r + 8 * hh][u * 16 + l16] = (__bf16)pf[u][r];

    Frag ap[2];
#pragma unroll
    for (int ks = 0; ks < 2; ++ks) {
      ap[ks].q[0] = *(const uint4*)&Ps[wave][l16][ks * 32 + hh * 8];
      ap[ks].q[1] = *(const uint4*)&Ps[wave][l16][ks * 32 + 16 + hh * 8];
    }
#pragma unroll
    for (int ks = 0; ks < 2; ++ks)
#pragma unroll
      for (int u = 0; u < 4; ++u) {
        Frag bv;                                     // kdim=key contiguous in Vs row
        bv.q[0] = *(const uint4*)&Vs[buf][u * 16 + l16][ks * 32 + hh * 8];
        bv.q[1] = *(const uint4*)&Vs[buf][u * 16 + l16][ks * 32 + 16 + hh * 8];
        o[u] = __builtin_amdgcn_wmma_f32_16x16x32_bf16(
            false, ap[ks].v, false, bv.v, (short)0, o[u], false, false);
      }

    tile_copy_wait();
    __syncthreads();
  }

#pragma unroll
  for (int u = 0; u < 4; ++u)
#pragma unroll
    for (int r = 0; r < 8; ++r) {
      int srow = q0 + wave * 16 + r + 8 * hh;
      attended[((size_t)b * cS + srow) * cH + h * cHD + u * 16 + l16] =
          (__bf16)(o[u][r] / lrow[r]);
    }
}

// ---------------------------------------------------------------------------
// Kernel 3: output projection + bias + residual -> y fp32 (d_out)
// ---------------------------------------------------------------------------
__global__ __launch_bounds__(256) void out_gemm(
    const __bf16* __restrict__ Att, const __bf16* __restrict__ WoutT,
    const float* __restrict__ bout, const float* __restrict__ x,
    float* __restrict__ y)
{
  __shared__ __align__(16) __bf16 As[2][128][40];
  __shared__ __align__(16) __bf16 Bt[2][128][40];

  const int tid  = threadIdx.x;
  const int wave = tid >> 5, lane = tid & 31;
  const int hh   = lane >> 4, l16 = lane & 15;
  const int wm   = wave >> 1, wn  = wave & 1;
  const int m0   = blockIdx.x * 128, n0 = blockIdx.y * 128;

  auto load_tiles = [&](int buf, int k0) {
#pragma unroll
    for (int i = 0; i < 2; ++i) {
      int idx = tid + i * 256;
      int r = idx >> 2, c8 = idx & 3;
      tile_copy_b128(&As[buf][r][c8 * 8], Att + (size_t)(m0 + r) * cH + k0 + c8 * 8);
      tile_copy_b128(&Bt[buf][r][c8 * 8], WoutT + (size_t)(n0 + r) * cH + k0 + c8 * 8);
    }
  };

  f32x8 acc[2][4];
#pragma unroll
  for (int t = 0; t < 2; ++t)
#pragma unroll
    for (int u = 0; u < 4; ++u) acc[t][u] = zero_f32x8();

  load_tiles(0, 0);
  tile_copy_wait();
  __syncthreads();

  for (int kt = 0; kt < cH / 32; ++kt) {
    const int buf = kt & 1;
    if (kt + 1 < cH / 32) load_tiles(buf ^ 1, (kt + 1) * 32);

    Frag a[2], bb[4];
#pragma unroll
    for (int t = 0; t < 2; ++t) {
      int row = wm * 32 + t * 16 + l16;
      a[t].q[0] = *(const uint4*)&As[buf][row][hh * 8];
      a[t].q[1] = *(const uint4*)&As[buf][row][16 + hh * 8];
    }
#pragma unroll
    for (int u = 0; u < 4; ++u) {
      int col = wn * 64 + u * 16 + l16;
      bb[u].q[0] = *(const uint4*)&Bt[buf][col][hh * 8];
      bb[u].q[1] = *(const uint4*)&Bt[buf][col][16 + hh * 8];
    }
#pragma unroll
    for (int t = 0; t < 2; ++t)
#pragma unroll
      for (int u = 0; u < 4; ++u)
        acc[t][u] = __builtin_amdgcn_wmma_f32_16x16x32_bf16(
            false, a[t].v, false, bb[u].v, (short)0, acc[t][u], false, false);

    tile_copy_wait();
    __syncthreads();
  }

#pragma unroll
  for (int t = 0; t < 2; ++t)
#pragma unroll
    for (int u = 0; u < 4; ++u) {
      int n_g = n0 + wn * 64 + u * 16 + l16;
      float bias = bout[n_g];
#pragma unroll
      for (int r = 0; r < 8; ++r) {
        int m_g = m0 + wm * 32 + t * 16 + r + 8 * hh;
        size_t off = (size_t)m_g * cH + n_g;
        y[off] = acc[t][u][r] + bias + x[off];
      }
    }
}

// ---------------------------------------------------------------------------
// Kernel 4: LayerNorm in place over each H=1024 row of d_out.
// ---------------------------------------------------------------------------
__global__ __launch_bounds__(256) void layernorm_inplace(
    float* __restrict__ y, const float* __restrict__ gamma,
    const float* __restrict__ beta)
{
  __shared__ float red[256];
  const int row = blockIdx.x, tid = threadIdx.x;
  float v[4];
  float s = 0.0f;
#pragma unroll
  for (int i = 0; i < 4; ++i) {
    v[i] = y[(size_t)row * cH + tid + i * 256];
    s += v[i];
  }
  red[tid] = s;
  __syncthreads();
  for (int off = 128; off > 0; off >>= 1) {
    if (tid < off) red[tid] += red[tid + off];
    __syncthreads();
  }
  const float mu = red[0] * (1.0f / cH);
  __syncthreads();
  float d2 = 0.0f;
#pragma unroll
  for (int i = 0; i < 4; ++i) { float d = v[i] - mu; d2 += d * d; }
  red[tid] = d2;
  __syncthreads();
  for (int off = 128; off > 0; off >>= 1) {
    if (tid < off) red[tid] += red[tid + off];
    __syncthreads();
  }
  const float rstd = rsqrtf(red[0] * (1.0f / cH) + 1e-5f);
#pragma unroll
  for (int i = 0; i < 4; ++i) {
    int c = tid + i * 256;
    y[(size_t)row * cH + c] = (v[i] - mu) * rstd * gamma[c] + beta[c];
  }
}

// ---------------------------------------------------------------------------
extern "C" void kernel_launch(void* const* d_in, const int* in_sizes, int n_in,
                              void* d_out, int out_size, void* d_ws, size_t ws_size,
                              hipStream_t stream)
{
  (void)in_sizes; (void)n_in; (void)out_size; (void)ws_size;
  const float* x     = (const float*)d_in[0];
  const int*   mask  = (const int*)d_in[1];
  const float* Wqkv  = (const float*)d_in[2];
  const float* bqkv  = (const float*)d_in[3];
  const float* Wout  = (const float*)d_in[4];
  const float* bout  = (const float*)d_in[5];
  const float* gamma = (const float*)d_in[6];
  const float* beta  = (const float*)d_in[7];
  float* out = (float*)d_out;

  // Workspace (bf16 elements): 24M elems = 48 MB
  __bf16* xb    = (__bf16*)d_ws;                       // [4096][1024]
  __bf16* WqkvT = xb    + (size_t)cM * cH;             // [3072][1024]
  __bf16* WoutT = WqkvT + (size_t)c3H * cH;            // [1024][1024]
  __bf16* Q     = WoutT + (size_t)cH * cH;             // [B][NH][S][HD]
  __bf16* K     = Q + (size_t)cB * cNH * cS * cHD;
  __bf16* Vt    = K + (size_t)cB * cNH * cS * cHD;     // [B][NH][HD][S]
  __bf16* At    = Vt + (size_t)cB * cNH * cS * cHD;    // [4096][1024]

  cvt_bf16<<<dim3((cM * cH) / 1024), 256, 0, stream>>>(x, xb);
  cvt_t_bf16<<<dim3(c3H / 32, cH / 32), 256, 0, stream>>>(Wqkv, WqkvT, cH, c3H);
  cvt_t_bf16<<<dim3(cH / 32, cH / 32), 256, 0, stream>>>(Wout, WoutT, cH, cH);

  qkv_gemm<<<dim3(cM / 128, c3H / 128), 256, 0, stream>>>(xb, WqkvT, bqkv, Q, K, Vt);
  flash_attn<<<dim3(cS / 64, cNH, cB), 128, 0, stream>>>(Q, K, Vt, mask, At);
  out_gemm<<<dim3(cM / 128, cH / 128), 256, 0, stream>>>(At, WoutT, bout, x, out);
  layernorm_inplace<<<dim3(cM), 256, 0, stream>>>(out, gamma, beta);
}
